// MoE_37211596653141
// MI455X (gfx1250) — matrix operand order repaired and checked by reading, not compiled, not run on hardware
//
#include <hip/hip_runtime.h>
#include <math.h>

// MoE: B=4, S=2048, D=1024, H=4096, E=8, TOP_K=2
// T = B*S = 8192 tokens. out[T][D] = sum_k g_k * ( relu(x W1[e_k] + b1) W2[e_k] + b2 )

#define T_TOK   8192
#define DD      1024
#define HH      4096
#define EE      8
#define HCHUNK  128
#define MTILE   32

typedef __attribute__((ext_vector_type(16))) _Float16 v16h;
typedef __attribute__((ext_vector_type(8)))  _Float16 v8h;
typedef __attribute__((ext_vector_type(8)))  float    v8f;

// workspace layout (bytes)
#define WS_GATES 256u
#define WS_LIST  (256u + 65536u)                    // E*T ints follow gates
#define WS_XH    (1u << 20)                         // x as f16          [T][D]     16 MB
#define WS_W1T   (WS_XH + 16777216u)                // W1^T as f16       [E][H][D]  64 MB
#define WS_W2T   (WS_W1T + 67108864u)               // W2^T as f16       [E][D][H]  64 MB
#define WS_END   (WS_W2T + 67108864u)               // ~145 MB total

// ---------------- kernel 0: zero output + counters ----------------
__global__ __launch_bounds__(256) void moe_zero(float* __restrict__ out, int* __restrict__ cnt) {
    size_t i = (size_t)blockIdx.x * blockDim.x + threadIdx.x;   // 8192*256 = 2M float4
    float4 z; z.x = 0.f; z.y = 0.f; z.z = 0.f; z.w = 0.f;
    ((float4*)out)[i] = z;
    if (blockIdx.x == 0 && threadIdx.x < EE) cnt[threadIdx.x] = 0;
}

// ---------------- kernel 1: router ----------------
__global__ __launch_bounds__(128) void moe_router(const float* __restrict__ x,
                                                  const float* __restrict__ Wr,
                                                  const float* __restrict__ br,
                                                  float* __restrict__ gates,
                                                  int* __restrict__ cnt,
                                                  int* __restrict__ list) {
    __shared__ float part[128][EE];
    const int tok = blockIdx.x;
    const int tid = threadIdx.x;
    const float* xr = x + (size_t)tok * DD;

    float acc[EE];
#pragma unroll
    for (int e = 0; e < EE; ++e) acc[e] = 0.f;
    for (int d = tid; d < DD; d += 128) {
        float xv = xr[d];
        const float* w = Wr + (size_t)d * EE;
#pragma unroll
        for (int e = 0; e < EE; ++e) acc[e] += xv * w[e];
    }
#pragma unroll
    for (int e = 0; e < EE; ++e) part[tid][e] = acc[e];
    __syncthreads();
    for (int off = 64; off > 0; off >>= 1) {
        if (tid < off) {
#pragma unroll
            for (int e = 0; e < EE; ++e) part[tid][e] += part[tid + off][e];
        }
        __syncthreads();
    }
    if (tid == 0) {
        float l[EE];
#pragma unroll
        for (int e = 0; e < EE; ++e) l[e] = part[0][e] + br[e];
        int i0 = 0;
#pragma unroll
        for (int e = 1; e < EE; ++e) if (l[e] > l[i0]) i0 = e;
        int i1 = -1;
#pragma unroll
        for (int e = 0; e < EE; ++e) {
            if (e == i0) continue;
            if (i1 < 0 || l[e] > l[i1]) i1 = e;
        }
        float e1 = __expf(l[i1] - l[i0]);     // e0 = 1
        float inv = 1.0f / (1.0f + e1);
        gates[tok * 2 + 0] = inv;
        gates[tok * 2 + 1] = e1 * inv;
        int p0 = atomicAdd(&cnt[i0], 1); list[i0 * T_TOK + p0] = tok * 2 + 0;
        int p1 = atomicAdd(&cnt[i1], 1); list[i1 * T_TOK + p1] = tok * 2 + 1;
    }
}

// ---------------- precision pre-pass: f32 -> f16 (and transpose weights) ----
__global__ __launch_bounds__(256) void moe_cvt_x(const float* __restrict__ x,
                                                 _Float16* __restrict__ xh) {
    size_t i = ((size_t)blockIdx.x * 256 + threadIdx.x) * 8;    // 4096 blocks
    float4 a = *(const float4*)(x + i);
    float4 b = *(const float4*)(x + i + 4);
    v8h o;
    o[0]=(_Float16)a.x; o[1]=(_Float16)a.y; o[2]=(_Float16)a.z; o[3]=(_Float16)a.w;
    o[4]=(_Float16)b.x; o[5]=(_Float16)b.y; o[6]=(_Float16)b.z; o[7]=(_Float16)b.w;
    *(v8h*)(xh + i) = o;
}

// src: [E][R][C] f32  ->  dst: [E][C][R] f16   (32x32 LDS tile transpose)
__global__ __launch_bounds__(256) void moe_cvt_t(const float* __restrict__ src,
                                                 _Float16* __restrict__ dst,
                                                 int R, int C) {
    __shared__ float t[32][33];
    const int e  = blockIdx.z;
    const size_t so = (size_t)e * R * C;
    const int r0 = blockIdx.y * 32;
    const int c0 = blockIdx.x * 32;
    const int tx = threadIdx.x & 31;
    const int ty = threadIdx.x >> 5;   // 0..7
#pragma unroll
    for (int i = 0; i < 4; ++i) {
        int r = ty + 8 * i;
        t[r][tx] = src[so + (size_t)(r0 + r) * C + c0 + tx];
    }
    __syncthreads();
#pragma unroll
    for (int i = 0; i < 4; ++i) {
        int c = ty + 8 * i;
        dst[so + (size_t)(c0 + c) * R + r0 + tx] = (_Float16)t[tx][c];
    }
}

// ---------------- async copy helpers (CDNA5 global -> LDS, ASYNCcnt) --------
__device__ __forceinline__ void async_b128(unsigned lds, const void* g) {
    asm volatile("global_load_async_to_lds_b128 %0, %1, off"
                 :: "v"(lds), "v"((unsigned long long)(size_t)g) : "memory");
}
__device__ __forceinline__ void async_b64(unsigned lds, const void* g) {
    asm volatile("global_load_async_to_lds_b64 %0, %1, off"
                 :: "v"(lds), "v"((unsigned long long)(size_t)g) : "memory");
}
__device__ __forceinline__ void async_wait0() {
    asm volatile("s_wait_asynccnt 0x0" ::: "memory");
}
__device__ __forceinline__ unsigned lds_off(const void* p) {
    return (unsigned)(size_t)p;   // flat LDS address: low 32 bits = LDS byte offset
}

// =================== kernel 2a: fused FFN, f16 weights + async staging =======
#define APITCH   1032   // 32 x 1032 f16  = 66048 B  (row stride 2064 B: 16B aligned, banks 4m)
#define W1PITCH  36     // 128 x 36 f16   = 9216 B   ([hcol][k])
#define HPITCH   132    // 32 x 132 f16   = 8448 B
#define W2PITCH  36     // 1024 x 36 f16  = 73728 B  ([dcol][k])

__global__ __launch_bounds__(256) void moe_ffn_f16(const _Float16* __restrict__ xh,
                                                   const _Float16* __restrict__ W1t, // [E][H][D]
                                                   const float* __restrict__ b1,
                                                   const _Float16* __restrict__ W2t, // [E][D][H]
                                                   const float* __restrict__ b2,
                                                   const float* __restrict__ gates,
                                                   const int* __restrict__ cnt,
                                                   const int* __restrict__ list,
                                                   float* __restrict__ out) {
    const int e    = blockIdx.y;
    const int tile = blockIdx.x;
    const int ne   = cnt[e];
    if (tile * MTILE >= ne) return;

    __shared__ __attribute__((aligned(16))) _Float16 sA [MTILE * APITCH];
    __shared__ __attribute__((aligned(16))) _Float16 sW1[128 * W1PITCH];
    __shared__ __attribute__((aligned(16))) _Float16 sH [MTILE * HPITCH];
    __shared__ __attribute__((aligned(16))) _Float16 sW2[1024 * W2PITCH];
    __shared__ int   sTok[MTILE];
    __shared__ float sGate[MTILE];

    const int tid  = threadIdx.x;
    const int lane = tid & 31;
    const int wv   = tid >> 5;
    const int mr   = wv & 1;          // row tile (16 rows)
    const int grp  = wv >> 1;         // column group
    const int half = lane >> 4;
    const int n    = lane & 15;

    if (tid < MTILE) {
        int idx = tile * MTILE + tid;
        if (idx < ne) {
            int rec = list[e * T_TOK + idx];
            sTok[tid]  = rec >> 1;
            sGate[tid] = gates[rec];
        } else {
            int rec0 = list[e * T_TOK + tile * MTILE];
            sTok[tid]  = rec0 >> 1;
            sGate[tid] = 0.0f;
        }
    }
    __syncthreads();

    // ---- A tile [32 x 1024] f16: async gather rows into LDS ----
    {
        const int th   = tid >> 7;      // 0/1
        const int cseg = tid & 127;     // 16B segment (8 f16) within a row
#pragma unroll
        for (int i = 0; i < 16; ++i) {
            int row = 2 * i + th;
            const _Float16* g = xh + (size_t)sTok[row] * DD + cseg * 8;
            async_b128(lds_off(&sA[row * APITCH + cseg * 8]), g);
        }
        async_wait0();
    }
    __syncthreads();

    v8f yacc[16];
    {
        v8f z = {0.f, 0.f, 0.f, 0.f, 0.f, 0.f, 0.f, 0.f};
#pragma unroll
        for (int t = 0; t < 16; ++t) yacc[t] = z;
    }

    const _Float16* W1e = W1t + (size_t)e * HH * DD;   // [h][d]
    const _Float16* W2e = W2t + (size_t)e * DD * HH;   // [d][h]

    for (int hc = 0; hc < HH; hc += HCHUNK) {
        // ============ GEMM1: h[32][128] = relu(A @ W1[:,hc:hc+128] + b1) ============
        v8f hacc[2];
        {
            v8f z = {0.f, 0.f, 0.f, 0.f, 0.f, 0.f, 0.f, 0.f};
            hacc[0] = z; hacc[1] = z;
        }
        for (int k0 = 0; k0 < DD; k0 += 32) {
            __syncthreads();   // previous sW1 readers done
            {   // async stage W1t tile: hcol 0..127, k(d) 0..31 -> sW1[hcol][k]
#pragma unroll
                for (int i = 0; i < 4; ++i) {
                    int chunk = i * 256 + tid;      // 1024 chunks of 8B (4 f16)
                    int hcol  = chunk >> 3;
                    int seg   = chunk & 7;
                    const _Float16* g = W1e + (size_t)(hc + hcol) * DD + k0 + seg * 4;
                    async_b64(lds_off(&sW1[hcol * W1PITCH + seg * 4]), g);
                }
                async_wait0();
            }
            __syncthreads();
            // A fragment (contiguous k = 16*half + j, same order as B)
            v16h af;
            {
                const _Float16* ap = &sA[(16 * mr + n) * APITCH + k0 + 16 * half];
#pragma unroll
                for (int j = 0; j < 8; ++j)  af[j] = ap[j];
#pragma unroll
                for (int j = 8; j < 16; ++j) af[j] = ap[j];
            }
#pragma unroll
            for (int t = 0; t < 2; ++t) {
                v16h bf;
                const _Float16* bp = &sW1[(32 * grp + 16 * t + n) * W1PITCH + 16 * half];
#pragma unroll
                for (int j = 0; j < 16; ++j) bf[j] = bp[j];
                hacc[t] = __builtin_amdgcn_wmma_f32_16x16x32_f16(false, af, false, bf,
                                                                 (short)0, hacc[t], false, false);
            }
        }

        __syncthreads();
#pragma unroll
        for (int t = 0; t < 2; ++t) {
            float b1v = b1[(size_t)e * HH + hc + 32 * grp + 16 * t + n];
#pragma unroll
            for (int r = 0; r < 8; ++r) {
                int m = 16 * mr + r + 8 * half;
                float hv = hacc[t][r] + b1v;
                hv = fmaxf(hv, 0.0f);
                sH[m * HPITCH + 32 * grp + 16 * t + n] = (_Float16)hv;
            }
        }
        __syncthreads();

        // ============ GEMM2: y[32][1024] += h @ W2[hc:hc+128, :] ====================
        for (int k2 = 0; k2 < HCHUNK; k2 += 32) {
            {   // async stage W2t tile: dcol 0..1023, k(h) 0..31 -> sW2[dcol][k]
#pragma unroll
                for (int i = 0; i < 32; ++i) {
                    int chunk = i * 256 + tid;      // 8192 chunks of 8B
                    int dcol  = chunk >> 3;
                    int seg   = chunk & 7;
                    const _Float16* g = W2e + (size_t)dcol * HH + hc + k2 + seg * 4;
                    async_b64(lds_off(&sW2[dcol * W2PITCH + seg * 4]), g);
                }
                async_wait0();
            }
            __syncthreads();
            v16h a2;
            {
                const _Float16* hp = &sH[(16 * mr + n) * HPITCH + k2 + 16 * half];
#pragma unroll
                for (int j = 0; j < 8; ++j)  a2[j] = hp[j];
#pragma unroll
                for (int j = 8; j < 16; ++j) a2[j] = hp[j];
            }
#pragma unroll
            for (int t = 0; t < 16; ++t) {
                v16h bf;
                const _Float16* bp = &sW2[(256 * grp + 16 * t + n) * W2PITCH + 16 * half];
#pragma unroll
                for (int j = 0; j < 16; ++j) bf[j] = bp[j];
                yacc[t] = __builtin_amdgcn_wmma_f32_16x16x32_f16(false, a2, false, bf,
                                                                 (short)0, yacc[t], false, false);
            }
            __syncthreads();
        }
    }

    // ---- epilogue: (y + b2) * gate, scatter-add ----
#pragma unroll
    for (int t = 0; t < 16; ++t) {
        int col = 256 * grp + 16 * t + n;
        float b2v = b2[(size_t)e * DD + col];
#pragma unroll
        for (int r = 0; r < 8; ++r) {
            int m = 16 * mr + r + 8 * half;
            float gv = sGate[m];
            if (gv != 0.0f) {
                float val = (yacc[t][r] + b2v) * gv;
                atomicAdd(out + (size_t)sTok[m] * DD + col, val);
            }
        }
    }
}

// =================== kernel 2b: fallback fused FFN (f32 weights, small ws) ===
__global__ __launch_bounds__(256) void moe_ffn(const float* __restrict__ x,
                                               const float* __restrict__ W1,
                                               const float* __restrict__ b1,
                                               const float* __restrict__ W2,
                                               const float* __restrict__ b2,
                                               const float* __restrict__ gates,
                                               const int* __restrict__ cnt,
                                               const int* __restrict__ list,
                                               float* __restrict__ out) {
    const int e    = blockIdx.y;
    const int tile = blockIdx.x;
    const int ne   = cnt[e];
    if (tile * MTILE >= ne) return;

    __shared__ __attribute__((aligned(16))) _Float16 sA [MTILE * APITCH];
    __shared__ __attribute__((aligned(16))) _Float16 sW1[128 * W1PITCH];
    __shared__ __attribute__((aligned(16))) _Float16 sH [MTILE * HPITCH];
    __shared__ __attribute__((aligned(16))) _Float16 sW2[1024 * W2PITCH];
    __shared__ int   sTok[MTILE];
    __shared__ float sGate[MTILE];

    const int tid  = threadIdx.x;
    const int lane = tid & 31;
    const int wv   = tid >> 5;
    const int mr   = wv & 1;
    const int grp  = wv >> 1;
    const int half = lane >> 4;
    const int n    = lane & 15;

    if (tid < MTILE) {
        int idx = tile * MTILE + tid;
        if (idx < ne) {
            int rec = list[e * T_TOK + idx];
            sTok[tid]  = rec >> 1;
            sGate[tid] = gates[rec];
        } else {
            int rec0 = list[e * T_TOK + tile * MTILE];
            sTok[tid]  = rec0 >> 1;
            sGate[tid] = 0.0f;
        }
    }
    __syncthreads();

    {
        int row = tid >> 3;
        int li  = tid & 7;
        const float* xr = x + (size_t)sTok[row] * DD;
        _Float16* ar = &sA[row * APITCH];
#pragma unroll
        for (int i = 0; i < 32; ++i) {
            int c4 = (li + 8 * i) * 4;
            float4 v = *(const float4*)(xr + c4);
            ar[c4 + 0] = (_Float16)v.x; ar[c4 + 1] = (_Float16)v.y;
            ar[c4 + 2] = (_Float16)v.z; ar[c4 + 3] = (_Float16)v.w;
        }
    }

    v8f yacc[16];
    {
        v8f z = {0.f, 0.f, 0.f, 0.f, 0.f, 0.f, 0.f, 0.f};
#pragma unroll
        for (int t = 0; t < 16; ++t) yacc[t] = z;
    }

    const float* W1e = W1 + (size_t)e * DD * HH;
    const float* W2e = W2 + (size_t)e * HH * DD;

    for (int hc = 0; hc < HH; hc += HCHUNK) {
        v8f hacc[2];
        {
            v8f z = {0.f, 0.f, 0.f, 0.f, 0.f, 0.f, 0.f, 0.f};
            hacc[0] = z; hacc[1] = z;
        }
        for (int k0 = 0; k0 < DD; k0 += 32) {
            __syncthreads();
            {
                int f4 = tid & 31;
                int r8 = tid >> 5;
#pragma unroll
                for (int rp = 0; rp < 4; ++rp) {
                    int r = rp * 8 + r8;
                    const float* src = W1e + (size_t)(k0 + r) * HH + hc + f4 * 4;
                    float4 v = *(const float4*)src;
                    if (k0 + 32 < DD)
                        __builtin_prefetch(src + (size_t)32 * HH, 0, 0);
                    sW1[(f4 * 4 + 0) * W1PITCH + r] = (_Float16)v.x;
                    sW1[(f4 * 4 + 1) * W1PITCH + r] = (_Float16)v.y;
                    sW1[(f4 * 4 + 2) * W1PITCH + r] = (_Float16)v.z;
                    sW1[(f4 * 4 + 3) * W1PITCH + r] = (_Float16)v.w;
                }
            }
            __syncthreads();
            v16h af;
            {
                const _Float16* ap = &sA[(16 * mr + n) * APITCH + k0 + 16 * half];
#pragma unroll
                for (int j = 0; j < 8; ++j)  af[j] = ap[j];
#pragma unroll
                for (int j = 8; j < 16; ++j) af[j] = ap[j];
            }
#pragma unroll
            for (int t = 0; t < 2; ++t) {
                v16h bf;
                const _Float16* bp = &sW1[(32 * grp + 16 * t + n) * W1PITCH + 16 * half];
#pragma unroll
                for (int j = 0; j < 16; ++j) bf[j] = bp[j];
                hacc[t] = __builtin_amdgcn_wmma_f32_16x16x32_f16(false, af, false, bf,
                                                                 (short)0, hacc[t], false, false);
            }
        }

        __syncthreads();
#pragma unroll
        for (int t = 0; t < 2; ++t) {
            float b1v = b1[(size_t)e * HH + hc + 32 * grp + 16 * t + n];
#pragma unroll
            for (int r = 0; r < 8; ++r) {
                int m = 16 * mr + r + 8 * half;
                float hv = hacc[t][r] + b1v;
                hv = fmaxf(hv, 0.0f);
                sH[m * HPITCH + 32 * grp + 16 * t + n] = (_Float16)hv;
            }
        }
        __syncthreads();

        for (int k2 = 0; k2 < HCHUNK; k2 += 32) {
            {
                int c4 = tid;
#pragma unroll 4
                for (int r = 0; r < 32; ++r) {
                    const float* src = W2e + (size_t)(hc + k2 + r) * DD + c4 * 4;
                    float4 v = *(const float4*)src;
                    if (k2 + 32 < HCHUNK)
                        __builtin_prefetch(src + (size_t)32 * DD, 0, 0);
                    sW2[(c4 * 4 + 0) * W2PITCH + r] = (_Float16)v.x;
                    sW2[(c4 * 4 + 1) * W2PITCH + r] = (_Float16)v.y;
                    sW2[(c4 * 4 + 2) * W2PITCH + r] = (_Float16)v.z;
                    sW2[(c4 * 4 + 3) * W2PITCH + r] = (_Float16)v.w;
                }
            }
            __syncthreads();
            v16h a2;
            {
                const _Float16* hp = &sH[(16 * mr + n) * HPITCH + k2 + 16 * half];
#pragma unroll
                for (int j = 0; j < 8; ++j)  a2[j] = hp[j];
#pragma unroll
                for (int j = 8; j < 16; ++j) a2[j] = hp[j];
            }
#pragma unroll
            for (int t = 0; t < 16; ++t) {
                v16h bf;
                const _Float16* bp = &sW2[(256 * grp + 16 * t + n) * W2PITCH + 16 * half];
#pragma unroll
                for (int j = 0; j < 16; ++j) bf[j] = bp[j];
                yacc[t] = __builtin_amdgcn_wmma_f32_16x16x32_f16(false, a2, false, bf,
                                                                 (short)0, yacc[t], false, false);
            }
            __syncthreads();
        }
    }

#pragma unroll
    for (int t = 0; t < 16; ++t) {
        int col = 256 * grp + 16 * t + n;
        float b2v = b2[(size_t)e * DD + col];
#pragma unroll
        for (int r = 0; r < 8; ++r) {
            int m = 16 * mr + r + 8 * half;
            float gv = sGate[m];
            if (gv != 0.0f) {
                float val = (yacc[t][r] + b2v) * gv;
                atomicAdd(out + (size_t)sTok[m] * DD + col, val);
            }
        }
    }
}

// ---------------- host launcher ----------------
extern "C" void kernel_launch(void* const* d_in, const int* in_sizes, int n_in,
                              void* d_out, int out_size, void* d_ws, size_t ws_size,
                              hipStream_t stream) {
    (void)in_sizes; (void)n_in; (void)out_size;
    const float* x  = (const float*)d_in[0];
    const float* Wr = (const float*)d_in[1];
    const float* br = (const float*)d_in[2];
    const float* W1 = (const float*)d_in[3];
    const float* b1 = (const float*)d_in[4];
    const float* W2 = (const float*)d_in[5];
    const float* b2 = (const float*)d_in[6];
    float* out = (float*)d_out;

    char* ws = (char*)d_ws;
    int*   cnt   = (int*)ws;
    float* gates = (float*)(ws + WS_GATES);
    int*   list  = (int*)(ws + WS_LIST);

    moe_zero<<<8192, 256, 0, stream>>>(out, cnt);
    moe_router<<<T_TOK, 128, 0, stream>>>(x, Wr, br, gates, cnt, list);

    if (ws_size >= (size_t)WS_END) {
        // fast path: pre-convert x/W1/W2 to f16 (weights transposed to B-layout),
        // then async-staged WMMA FFN (halves hot-loop HBM traffic).
        _Float16* xh  = (_Float16*)(ws + WS_XH);
        _Float16* W1t = (_Float16*)(ws + WS_W1T);
        _Float16* W2t = (_Float16*)(ws + WS_W2T);
        moe_cvt_x<<<4096, 256, 0, stream>>>(x, xh);
        dim3 g1(HH / 32, DD / 32, EE);
        moe_cvt_t<<<g1, 256, 0, stream>>>(W1, W1t, DD, HH);   // [D][H] -> [H][D]
        dim3 g2(DD / 32, HH / 32, EE);
        moe_cvt_t<<<g2, 256, 0, stream>>>(W2, W2t, HH, DD);   // [H][D] -> [D][H]
        dim3 grid(T_TOK / MTILE, EE);
        moe_ffn_f16<<<grid, 256, 0, stream>>>(xh, W1t, b1, W2t, b2, gates, cnt, list, out);
    } else {
        // fallback: stage f32 weights with conversion in-kernel
        dim3 grid(T_TOK / MTILE, EE);
        moe_ffn<<<grid, 256, 0, stream>>>(x, W1, b1, W2, b2, gates, cnt, list, out);
    }
}
